// SwinTransformerBlock_72756745994856
// MI455X (gfx1250) — matrix operand order repaired and checked by reading, not compiled, not run on hardware
//
#include <hip/hip_runtime.h>
#include <hip/hip_bf16.h>
#include <cmath>

typedef __bf16 bf16;
typedef __attribute__((ext_vector_type(16))) __bf16 bf16x16;
typedef __attribute__((ext_vector_type(8)))  __bf16 bf16x8;
typedef __attribute__((ext_vector_type(8)))  float  floatx8;

// ---------------- problem constants ----------------
constexpr int BB_    = 8;
constexpr int H_     = 128, W_ = 128, C_ = 256;
constexpr int HEADS_ = 8, WS_ = 8, SS_ = 4;
constexpr int N_     = WS_ * WS_;                 // 64 tokens per window
constexpr int HD_    = C_ / HEADS_;               // 32
constexpr int NW_    = (H_ / WS_) * (W_ / WS_);   // 256 windows per image
constexpr int NWIN   = BB_ * NW_;                 // 2048 windows total
constexpr int TOK    = BB_ * H_ * W_;             // 131072 tokens
constexpr float SCALE_ = 0.17677669529663687f;    // HD^-0.5

// ---------------- WMMA fragment helpers (CDNA5 16x16x32 bf16) ----------------
__device__ inline bf16x16 make16(bf16x8 lo, bf16x8 hi) {
  return __builtin_shufflevector(lo, hi, 0,1,2,3,4,5,6,7,8,9,10,11,12,13,14,15);
}

// A operand: 16x32 tile from row-major [16][>=32] source, leading dim ld.
// Lane half g: row = lane&15, K elements {g*8..g*8+7} and {16+g*8..16+g*8+7}.
__device__ inline bf16x16 load_a16(const bf16* base, int ld) {
  const int lane = threadIdx.x & 31;
  const int r = lane & 15, g = lane >> 4;
  const bf16* p = base + r * ld + g * 8;
  return make16(*(const bf16x8*)p, *(const bf16x8*)(p + 16));
}

// B operand for Y = X * M^T: B[k][n] = M[n][k], M row-major [16][>=32], ld.
// Lane half g: col n = lane&15, K elements {g*8..g*8+7} and {16+g*8..16+g*8+7}.
__device__ inline bf16x16 load_b16(const bf16* base, int ld) {
  const int lane = threadIdx.x & 31;
  const int n = lane & 15, g = lane >> 4;
  const bf16* p = base + n * ld + g * 8;
  return make16(*(const bf16x8*)p, *(const bf16x8*)(p + 16));
}

__device__ inline floatx8 wmma_bf16(bf16x16 a, bf16x16 b, floatx8 c) {
  return __builtin_amdgcn_wmma_f32_16x16x32_bf16(false, a, false, b, (short)0, c,
                                                 false, false);
}

// One wave computes a 32-row x 64-col strip of Y = X * W^T  (W row-major [n][KD]).
// 12 b128 loads feed 8 WMMAs per K-step (vs 10:4 for a 16-row strip).
template <int KD>
__device__ inline void gemm_strip32(const bf16* __restrict__ xrow,
                                    const bf16* __restrict__ wbase,
                                    floatx8 (&acc)[2][4]) {
  for (int kk = 0; kk < KD; kk += 32) {
    bf16x16 a0 = load_a16(xrow + kk, KD);
    bf16x16 a1 = load_a16(xrow + (size_t)16 * KD + kk, KD);
#pragma unroll
    for (int t = 0; t < 4; ++t) {
      bf16x16 b = load_b16(wbase + (size_t)(t * 16) * KD + kk, KD);
      acc[0][t] = wmma_bf16(a0, b, acc[0][t]);
      acc[1][t] = wmma_bf16(a1, b, acc[1][t]);
    }
  }
}

// ---------------- small utility kernels ----------------
__global__ __launch_bounds__(256) void f2bf_kernel(const float* __restrict__ s,
                                                   bf16* __restrict__ d, int n) {
  int i = blockIdx.x * 256 + threadIdx.x;
  if (i < n) d[i] = (bf16)s[i];
}

// LayerNorm (+ optional cyclic shift & window partition). One wave per token.
__global__ __launch_bounds__(256) void ln_kernel(const float* __restrict__ x,
                                                 const float* __restrict__ gw,
                                                 const float* __restrict__ gb,
                                                 bf16* __restrict__ out,
                                                 int shifted) {
  const int gt = blockIdx.x * 8 + (threadIdx.x >> 5);  // output row
  const int lane = threadIdx.x & 31;
  int src = gt;
  if (shifted) {
    int wid = gt >> 6, n = gt & 63;
    int b = wid >> 8, wi = wid & 255;
    int hh = ((wi >> 4) << 3) + (n >> 3);
    int ww = ((wi & 15) << 3) + (n & 7);
    hh = (hh + SS_) & (H_ - 1);
    ww = (ww + SS_) & (W_ - 1);
    src = (b << 14) + (hh << 7) + ww;
  }
  const float* px = x + (size_t)src * C_;
  const int c0 = lane * 8;
  float v[8];
  float s = 0.f, qq = 0.f;
#pragma unroll
  for (int i = 0; i < 8; ++i) { v[i] = px[c0 + i]; s += v[i]; qq += v[i] * v[i]; }
#pragma unroll
  for (int d = 16; d >= 1; d >>= 1) {
    s  += __shfl_xor(s, d, 32);
    qq += __shfl_xor(qq, d, 32);
  }
  const float mean = s * (1.f / C_);
  const float var  = qq * (1.f / C_) - mean * mean;
  const float rstd = rsqrtf(var + 1e-5f);
  bf16* po = out + (size_t)gt * C_;
#pragma unroll
  for (int i = 0; i < 8; ++i)
    po[c0 + i] = (bf16)((v[i] - mean) * rstd * gw[c0 + i] + gb[c0 + i]);
}

// ---------------- QKV projection: (128x64) per block, 128 rows = 2 windows -----------
__global__ __launch_bounds__(128) void qkv_kernel(const bf16* __restrict__ xw,
                                                  const bf16* __restrict__ wq,
                                                  const float* __restrict__ bqkv,
                                                  bf16* __restrict__ q,
                                                  bf16* __restrict__ k,
                                                  bf16* __restrict__ vt) {
  const int bx = blockIdx.x;   // 128-row block (2 windows)
  const int by = blockIdx.y;   // 64-col tile of 768
  const int wv = threadIdx.x >> 5;
  const int lane = threadIdx.x & 31;
  const int c16 = lane & 15, g = lane >> 4;
  const int row0 = bx * 128 + 32 * wv;
  floatx8 acc[2][4];
#pragma unroll
  for (int i = 0; i < 2; ++i)
#pragma unroll
    for (int t = 0; t < 4; ++t) acc[i][t] = {};
  gemm_strip32<256>(xw + (size_t)row0 * C_, wq + (size_t)(by * 64) * C_, acc);
#pragma unroll
  for (int t = 0; t < 4; ++t) {
    const int j = by * 64 + 16 * t + c16;
    const float bias = bqkv[j];
    const int three = j >> 8, head = (j >> 5) & 7, d = j & 31;
#pragma unroll
    for (int i = 0; i < 2; ++i) {
#pragma unroll
      for (int r = 0; r < 8; ++r) {
        int row = row0 + 16 * i + r + 8 * g;
        int wid = row >> 6, n = row & 63;
        float val = acc[i][t][r] + bias;
        size_t whb = (size_t)wid * HEADS_ + head;
        if (three == 0)      q [(whb * N_ + n) * HD_ + d] = (bf16)(val * SCALE_);
        else if (three == 1) k [(whb * N_ + n) * HD_ + d] = (bf16)val;
        else                 vt[(whb * HD_ + d) * N_ + n] = (bf16)val;  // V^T
      }
    }
  }
}

// ---------------- windowed attention, one block per (window, head) ----------
__device__ inline int regid(int h) {
  return (h < H_ - WS_) ? 0 : ((h < H_ - SS_) ? 1 : 2);
}

__global__ __launch_bounds__(128) void attn_kernel(const bf16* __restrict__ q,
                                                   const bf16* __restrict__ k,
                                                   const bf16* __restrict__ vt,
                                                   const float* __restrict__ rpb,
                                                   bf16* __restrict__ out) {
  const int wh   = blockIdx.x;      // window*HEADS + head
  const int wid  = wh >> 3;
  const int head = wh & 7;
  const int wv   = threadIdx.x >> 5;
  const int lane = threadIdx.x & 31;
  const int c16  = lane & 15, g = lane >> 4;

  const bf16* qh = q  + (size_t)wh * (N_ * HD_);
  const bf16* kh = k  + (size_t)wh * (N_ * HD_);
  const bf16* vh = vt + (size_t)wh * (HD_ * N_);

  __shared__ __align__(16) bf16 pbuf[4][16][64];

  // S = (scaled Q) K^T  : wave strip 16x64, K = HD = 32 -> 4 WMMAs
  floatx8 s[4];
#pragma unroll
  for (int t = 0; t < 4; ++t) s[t] = {};
  bf16x16 afrag = load_a16(qh + (size_t)(16 * wv) * HD_, HD_);
#pragma unroll
  for (int t = 0; t < 4; ++t)
    s[t] = wmma_bf16(afrag, load_b16(kh + (size_t)(16 * t) * HD_, HD_), s[t]);

  // relative position bias + shifted-window mask, computed on the fly
  const int wi = wid & (NW_ - 1);
  const int h0 = (wi >> 4) << 3;
  const int w0 = (wi & 15) << 3;
#pragma unroll
  for (int t = 0; t < 4; ++t) {
#pragma unroll
    for (int r = 0; r < 8; ++r) {
      int n = 16 * wv + r + 8 * g;
      int m = 16 * t + c16;
      int ni = n >> 3, nj = n & 7, mi = m >> 3, mj = m & 7;
      float bias = rpb[((ni - mi + 7) * 15 + (nj - mj + 7)) * HEADS_ + head];
      int rn = regid(h0 + ni) * 3 + regid(w0 + nj);
      int rm = regid(h0 + mi) * 3 + regid(w0 + mj);
      s[t][r] += bias + ((rn == rm) ? 0.f : -100.f);
    }
  }

  // softmax: row n = 16*wv + r + 8*g lives in one 16-lane half (xor 1..8)
#pragma unroll
  for (int r = 0; r < 8; ++r) {
    float mx = s[0][r];
#pragma unroll
    for (int t = 1; t < 4; ++t) mx = fmaxf(mx, s[t][r]);
#pragma unroll
    for (int d = 8; d >= 1; d >>= 1) mx = fmaxf(mx, __shfl_xor(mx, d, 32));
    float sum = 0.f;
#pragma unroll
    for (int t = 0; t < 4; ++t) { float e = __expf(s[t][r] - mx); s[t][r] = e; sum += e; }
#pragma unroll
    for (int d = 8; d >= 1; d >>= 1) sum += __shfl_xor(sum, d, 32);
    float inv = 1.f / sum;
#pragma unroll
    for (int t = 0; t < 4; ++t)
      pbuf[wv][r + 8 * g][16 * t + c16] = (bf16)(s[t][r] * inv);
  }

  // O = P V : P 16x64 strip (LDS), V^T rows = head dims -> 4 WMMAs
  floatx8 o[2];
  o[0] = {}; o[1] = {};
#pragma unroll
  for (int ks = 0; ks < 2; ++ks) {
    bf16x16 pa = load_a16(&pbuf[wv][0][32 * ks], 64);
#pragma unroll
    for (int dt = 0; dt < 2; ++dt) {
      bf16x16 vb = load_b16(vh + (size_t)(16 * dt) * N_ + 32 * ks, N_);
      o[dt] = wmma_bf16(pa, vb, o[dt]);
    }
  }
#pragma unroll
  for (int dt = 0; dt < 2; ++dt) {
#pragma unroll
    for (int r = 0; r < 8; ++r) {
      int n = 16 * wv + r + 8 * g;
      out[((size_t)wid * N_ + n) * C_ + head * HD_ + 16 * dt + c16] = (bf16)o[dt][r];
    }
  }
}

// ---------------- proj GEMM + unpartition + reverse shift + residual --------
__global__ __launch_bounds__(128) void proj_kernel(const bf16* __restrict__ ain,
                                                   const bf16* __restrict__ wp,
                                                   const float* __restrict__ bp,
                                                   const float* __restrict__ x0,
                                                   float* __restrict__ x1) {
  const int bx = blockIdx.x, by = blockIdx.y;
  const int wv = threadIdx.x >> 5, lane = threadIdx.x & 31;
  const int c16 = lane & 15, g = lane >> 4;
  const int row0 = bx * 128 + 32 * wv;
  floatx8 acc[2][4];
#pragma unroll
  for (int i = 0; i < 2; ++i)
#pragma unroll
    for (int t = 0; t < 4; ++t) acc[i][t] = {};
  gemm_strip32<256>(ain + (size_t)row0 * C_, wp + (size_t)(by * 64) * C_, acc);
#pragma unroll
  for (int t = 0; t < 4; ++t) {
    const int cc = by * 64 + 16 * t + c16;
    const float bias = bp[cc];
#pragma unroll
    for (int i = 0; i < 2; ++i) {
#pragma unroll
      for (int r = 0; r < 8; ++r) {
        int row = row0 + 16 * i + r + 8 * g;
        int wid = row >> 6, n = row & 63;
        int b = wid >> 8, wi = wid & 255;
        int hh = ((((wi >> 4) << 3) + (n >> 3)) + SS_) & (H_ - 1);
        int ww = ((((wi & 15) << 3) + (n & 7)) + SS_) & (W_ - 1);
        size_t tok = ((size_t)b << 14) + (hh << 7) + ww;
        x1[tok * C_ + cc] = x0[tok * C_ + cc] + acc[i][t][r] + bias;
      }
    }
  }
}

// ---------------- MLP ----------------
__global__ __launch_bounds__(128) void fc1_kernel(const bf16* __restrict__ h,
                                                  const bf16* __restrict__ w1,
                                                  const float* __restrict__ b1,
                                                  bf16* __restrict__ h1) {
  const int bx = blockIdx.x, by = blockIdx.y;
  const int wv = threadIdx.x >> 5, lane = threadIdx.x & 31;
  const int c16 = lane & 15, g = lane >> 4;
  const int row0 = bx * 128 + 32 * wv;
  floatx8 acc[2][4];
#pragma unroll
  for (int i = 0; i < 2; ++i)
#pragma unroll
    for (int t = 0; t < 4; ++t) acc[i][t] = {};
  gemm_strip32<256>(h + (size_t)row0 * C_, w1 + (size_t)(by * 64) * C_, acc);
#pragma unroll
  for (int t = 0; t < 4; ++t) {
    const int j = by * 64 + 16 * t + c16;
    const float bias = b1[j];
#pragma unroll
    for (int i = 0; i < 2; ++i) {
#pragma unroll
      for (int r = 0; r < 8; ++r) {
        size_t row = (size_t)row0 + 16 * i + r + 8 * g;
        float val = acc[i][t][r] + bias;
        val = 0.5f * val * (1.f + erff(val * 0.70710678118654752f));  // exact GELU
        h1[row * 1024 + j] = (bf16)val;
      }
    }
  }
}

__global__ __launch_bounds__(128) void fc2_kernel(const bf16* __restrict__ h1,
                                                  const bf16* __restrict__ w2,
                                                  const float* __restrict__ b2,
                                                  const float* __restrict__ x1,
                                                  float* __restrict__ out) {
  const int bx = blockIdx.x, by = blockIdx.y;
  const int wv = threadIdx.x >> 5, lane = threadIdx.x & 31;
  const int c16 = lane & 15, g = lane >> 4;
  const int row0 = bx * 128 + 32 * wv;
  floatx8 acc[2][4];
#pragma unroll
  for (int i = 0; i < 2; ++i)
#pragma unroll
    for (int t = 0; t < 4; ++t) acc[i][t] = {};
  gemm_strip32<1024>(h1 + (size_t)row0 * 1024, w2 + (size_t)(by * 64) * 1024, acc);
#pragma unroll
  for (int t = 0; t < 4; ++t) {
    const int j = by * 64 + 16 * t + c16;
    const float bias = b2[j];
#pragma unroll
    for (int i = 0; i < 2; ++i) {
#pragma unroll
      for (int r = 0; r < 8; ++r) {
        size_t row = (size_t)row0 + 16 * i + r + 8 * g;
        out[row * C_ + j] = x1[row * C_ + j] + acc[i][t][r] + bias;
      }
    }
  }
}

// ---------------- workspace layout (bytes) ----------------
constexpr size_t OFF_QKVW = 0;                         // 768*256 bf16
constexpr size_t OFF_PROJW = OFF_QKVW + 393216;        // 256*256 bf16
constexpr size_t OFF_FC1W  = OFF_PROJW + 131072;       // 1024*256 bf16
constexpr size_t OFF_FC2W  = OFF_FC1W + 524288;        // 256*1024 bf16
constexpr size_t OFF_A     = OFF_FC2W + 524288;        // xw -> attn_out ; h1 overlays A..VT
constexpr size_t OFF_Q     = OFF_A + 67108864;
constexpr size_t OFF_K     = OFF_Q + 67108864;
constexpr size_t OFF_VT    = OFF_K + 67108864;
constexpr size_t OFF_X1    = OFF_VT + 67108864;        // fp32 residual
constexpr size_t OFF_H     = OFF_X1 + 134217728;       // LN2 out bf16

extern "C" void kernel_launch(void* const* d_in, const int* in_sizes, int n_in,
                              void* d_out, int out_size, void* d_ws, size_t ws_size,
                              hipStream_t stream) {
  (void)in_sizes; (void)n_in; (void)out_size; (void)ws_size;
  const float* x      = (const float*)d_in[0];
  const float* n1w    = (const float*)d_in[1];
  const float* n1b    = (const float*)d_in[2];
  const float* qkv_w  = (const float*)d_in[3];
  const float* qkv_b  = (const float*)d_in[4];
  const float* rpb    = (const float*)d_in[5];
  const float* proj_w = (const float*)d_in[6];
  const float* proj_b = (const float*)d_in[7];
  const float* n2w    = (const float*)d_in[8];
  const float* n2b    = (const float*)d_in[9];
  const float* fc1_w  = (const float*)d_in[10];
  const float* fc1_b  = (const float*)d_in[11];
  const float* fc2_w  = (const float*)d_in[12];
  const float* fc2_b  = (const float*)d_in[13];

  char* ws = (char*)d_ws;
  bf16* qkv_wb  = (bf16*)(ws + OFF_QKVW);
  bf16* proj_wb = (bf16*)(ws + OFF_PROJW);
  bf16* fc1_wb  = (bf16*)(ws + OFF_FC1W);
  bf16* fc2_wb  = (bf16*)(ws + OFF_FC2W);
  bf16* xw      = (bf16*)(ws + OFF_A);     // LN1/shift/partition output
  bf16* attn_o  = (bf16*)(ws + OFF_A);     // reuse after xw is consumed
  bf16* h1      = (bf16*)(ws + OFF_A);     // reuse A..VT after attention
  bf16* qbuf    = (bf16*)(ws + OFF_Q);
  bf16* kbuf    = (bf16*)(ws + OFF_K);
  bf16* vtbuf   = (bf16*)(ws + OFF_VT);
  float* x1     = (float*)(ws + OFF_X1);
  bf16* hbuf    = (bf16*)(ws + OFF_H);

  // 0) weights -> bf16
  f2bf_kernel<<<(196608 + 255) / 256, 256, 0, stream>>>(qkv_w, qkv_wb, 196608);
  f2bf_kernel<<<(65536 + 255) / 256, 256, 0, stream>>>(proj_w, proj_wb, 65536);
  f2bf_kernel<<<(262144 + 255) / 256, 256, 0, stream>>>(fc1_w, fc1_wb, 262144);
  f2bf_kernel<<<(262144 + 255) / 256, 256, 0, stream>>>(fc2_w, fc2_wb, 262144);

  // 1) LN1 + shift + window partition
  ln_kernel<<<TOK / 8, 256, 0, stream>>>(x, n1w, n1b, xw, 1);
  // 2) QKV projection (128 rows x 64 cols per block)
  qkv_kernel<<<dim3(NWIN / 2, 12), 128, 0, stream>>>(xw, qkv_wb, qkv_b,
                                                     qbuf, kbuf, vtbuf);
  // 3) windowed attention (overwrites xw region with attn output)
  attn_kernel<<<NWIN * HEADS_, 128, 0, stream>>>(qbuf, kbuf, vtbuf, rpb, attn_o);
  // 4) output proj + unshift + residual
  proj_kernel<<<dim3(NWIN / 2, 4), 128, 0, stream>>>(attn_o, proj_wb, proj_b, x, x1);
  // 5) LN2
  ln_kernel<<<TOK / 8, 256, 0, stream>>>(x1, n2w, n2b, hbuf, 0);
  // 6) FC1 + GELU (h1 overlays dead xw/q/k/vt region)
  fc1_kernel<<<dim3(TOK / 128, 16), 128, 0, stream>>>(hbuf, fc1_wb, fc1_b, h1);
  // 7) FC2 + residual -> out
  fc2_kernel<<<dim3(TOK / 128, 4), 128, 0, stream>>>(h1, fc2_wb, fc2_b, x1,
                                                     (float*)d_out);
}
